// CRFasRNN_85779086835829
// MI455X (gfx1250) — matrix operand order repaired and checked by reading, not compiled, not run on hardware
//
#include <hip/hip_runtime.h>
#include <hip/hip_bf16.h>

#define N_PIX 9216
#define HH 96
#define WW 96
#define NC 21
#define KSPLIT 8                       // K-dim (column) split for occupancy
#define MCHUNK (N_PIX / KSPLIT)        // 1152 columns per wave

typedef __attribute__((ext_vector_type(2)))  float    v2f;
typedef __attribute__((ext_vector_type(8)))  float    v8f;
typedef __attribute__((ext_vector_type(16))) _Float16 v16h;

// ---------------------------------------------------------------------------
// Unary network: 3x3 SAME cross-correlation, NCHW / OIHW.  u layout: [N][32]
// ---------------------------------------------------------------------------
__global__ __launch_bounds__(256) void unary_kernel(
    const float* __restrict__ img,   // [3][96][96]
    const float* __restrict__ w,     // [21][3][3][3]
    const float* __restrict__ b,     // [21]
    float* __restrict__ u)           // [N][32]
{
    int n = blockIdx.x * blockDim.x + threadIdx.x;
    if (n >= N_PIX) return;
    int y = n / WW, x = n % WW;
    float acc[NC];
    for (int o = 0; o < NC; ++o) acc[o] = b[o];
    for (int i = 0; i < 3; ++i) {
        for (int dy = -1; dy <= 1; ++dy) {
            int yy = y + dy;
            if (yy < 0 || yy >= HH) continue;
            for (int dx = -1; dx <= 1; ++dx) {
                int xx = x + dx;
                if (xx < 0 || xx >= WW) continue;
                float v = img[(i * HH + yy) * WW + xx];
                for (int o = 0; o < NC; ++o)
                    acc[o] += v * w[((o * 3 + i) * 3 + (dy + 1)) * 3 + (dx + 1)];
            }
        }
    }
    for (int o = 0; o < NC; ++o) u[(size_t)n * 32 + o] = acc[o];
}

// ---------------------------------------------------------------------------
// Per-pixel Gaussian features, padded to 8 dims (zeros contribute nothing).
// feat layout: [2 kernels][N][8] f32 ; sq layout: [2][N]
// kernel 0 (spatial):   (x/3, y/3, 0,...)
// kernel 1 (bilateral): (x/160, y/160, r/3, g/3, b/3, 0,...)
// ---------------------------------------------------------------------------
__global__ __launch_bounds__(256) void feat_kernel(
    const float* __restrict__ img, float* __restrict__ feat, float* __restrict__ sq)
{
    int n = blockIdx.x * blockDim.x + threadIdx.x;
    if (n >= N_PIX) return;
    float x = (float)(n % WW), y = (float)(n / WW);

    float f0 = x * (1.0f / 3.0f), f1 = y * (1.0f / 3.0f);
    float* fp = feat + (size_t)n * 8;
    fp[0] = f0; fp[1] = f1; fp[2] = 0.f; fp[3] = 0.f;
    fp[4] = 0.f; fp[5] = 0.f; fp[6] = 0.f; fp[7] = 0.f;
    sq[n] = f0 * f0 + f1 * f1;

    float g0 = x * (1.0f / 160.0f), g1 = y * (1.0f / 160.0f);
    float r = img[0 * N_PIX + n] * (1.0f / 3.0f);
    float g = img[1 * N_PIX + n] * (1.0f / 3.0f);
    float bl = img[2 * N_PIX + n] * (1.0f / 3.0f);
    float* fb = feat + (size_t)(N_PIX + n) * 8;
    fb[0] = g0; fb[1] = g1; fb[2] = r; fb[3] = g; fb[4] = bl;
    fb[5] = 0.f; fb[6] = 0.f; fb[7] = 0.f;
    sq[N_PIX + n] = g0 * g0 + g1 * g1 + r * r + g * g + bl * bl;
}

// ---------------------------------------------------------------------------
// Fused Gaussian-kernel generation + filtering, K-split for occupancy.
// One wave: 16-row tile of K (rows n0..n0+15), kernel kidx, columns
// [split*MCHUNK, (split+1)*MCHUNK).  Per 32-column chunk:
//   1. Gram tile G (16x32) via v_wmma_f32_16x16x4_f32 (k=4 steps, f32 exact).
//   2. K = exp(-0.5*max(|fn|^2+|fm|^2-2G, 0)) on C-layout regs -> LDS tile.
//   3. Re-read as f16 A-matrix (16x32 layout, ISA 7.12.2).
//   4. Two v_wmma_f32_16x16x32_f16: class tiles 0-15 / 16-31 of S^T.
// s channel 21 == 1.0 so output channel 21 accumulates the row normalizer.
// Partial sums -> filt[kidx][split][n][32]; reduced in pointwise_kernel.
// All column-stream pointers are hoisted & advanced by constants so the hot
// loop uses immediate-offset vector loads only.
// ---------------------------------------------------------------------------
__global__ __launch_bounds__(32) void filter_kernel(
    const float*    __restrict__ feat,  // [2][N][8]
    const float*    __restrict__ sq,    // [2][N]
    const _Float16* __restrict__ sT,    // [32][N]  class-major
    float*          __restrict__ filt)  // [2][KSPLIT][N][32]
{
    const int kidx  = blockIdx.y;          // 0 = spatial, 1 = bilateral
    const int split = blockIdx.z;
    const int n0    = blockIdx.x * 16;
    const int lane  = threadIdx.x;         // 0..31
    const int lh    = lane >> 4;           // lane half
    const int ll    = lane & 15;

    const float* F  = feat + (size_t)kidx * N_PIX * 8;
    const float* SQ = sq + (size_t)kidx * N_PIX;
    float* OUT      = filt + ((size_t)(kidx * KSPLIT + split) * N_PIX) * 32;

    const int mlo = split * MCHUNK;
    const int mhi = mlo + MCHUNK;

    __shared__ float tile[16][32];

    // Loop-invariant row features: A-matrix of the Gram WMMA (f32 16x4 layout:
    // lane l holds row M=l%16, K slots {2*(l/16), 2*(l/16)+1}).
    v2f arow0 = *(const v2f*)(F + (size_t)(n0 + ll) * 8 + 2 * lh);      // dims 0-3
    v2f arow1 = *(const v2f*)(F + (size_t)(n0 + ll) * 8 + 4 + 2 * lh);  // dims 4-7
    float sqn[8];
#pragma unroll
    for (int r = 0; r < 8; ++r) sqn[r] = SQ[n0 + r + 8 * lh];

    // Hoisted per-lane column-stream pointers (advance by constants).
    const float*    fcol = F + (size_t)(mlo + ll) * 8 + 2 * lh;
    const float*    sqp  = SQ + mlo + ll;
    const _Float16* bp0  = sT + (size_t)ll * N_PIX + mlo + 16 * lh;
    const _Float16* bp1  = sT + (size_t)(ll + 16) * N_PIX + mlo + 16 * lh;

    v8f acc0 = {};   // classes 0-15
    v8f acc1 = {};   // classes 16-31 (21 = normalizer)

    for (int m0 = mlo; m0 < mhi; m0 += 32) {
        // Prefetch the s-field stream ahead into the near (WGP) cache.
        if (m0 + 288 <= mhi) {
            __builtin_prefetch(bp0 + 256, 0, 3);
            __builtin_prefetch(bp1 + 256, 0, 3);
        }
#pragma unroll
        for (int half = 0; half < 2; ++half) {
            // B-matrix (f32 4x16): lane l holds col N=l%16, K slots 2*(l/16)+v
            v2f b0 = *(const v2f*)(fcol + half * 128);
            v8f g = {};
            g = __builtin_amdgcn_wmma_f32_16x16x4_f32(
                    false, arow0, false, b0, (short)0, g, false, false);
            if (kidx != 0) {
                v2f b1 = *(const v2f*)(fcol + half * 128 + 4);
                g = __builtin_amdgcn_wmma_f32_16x16x4_f32(
                        false, arow1, false, b1, (short)0, g, false, false);
            }
            float sqm = sqp[half * 16];
            // C-layout: VGPR r, lane l -> row r+8*(l/16), col l%16
#pragma unroll
            for (int r = 0; r < 8; ++r) {
                float d2 = fmaxf(sqn[r] + sqm - 2.0f * g[r], 0.0f);
                tile[r + 8 * lh][half * 16 + ll] = __expf(-0.5f * d2);
            }
        }
        // Cross-lane LDS hand-off: stores above feed other lanes' loads below.
        asm volatile("s_wait_dscnt 0" ::: "memory");

        // Rebuild K tile as f16 A-matrix (16x32): lane l = row l%16;
        // element e -> K = 16*(l/16)+e (contiguous, matches B)
        const int koff = 16 * lh;
        const float* rowp = &tile[ll][0];
        v16h a;
#pragma unroll
        for (int j = 0; j < 8; ++j) {
            a[j]     = (_Float16)rowp[koff + j];
            a[8 + j] = (_Float16)rowp[koff + 8 + j];
        }
        // B-matrix (f16 32x16): lane l = col l%16 (+16*tile), K = 16*(l/16)+e
        v16h bs0 = *(const v16h*)bp0;
        v16h bs1 = *(const v16h*)bp1;
        acc0 = __builtin_amdgcn_wmma_f32_16x16x32_f16(
                   false, a, false, bs0, (short)0, acc0, false, false);
        acc1 = __builtin_amdgcn_wmma_f32_16x16x32_f16(
                   false, a, false, bs1, (short)0, acc1, false, false);
        // Keep next iteration's LDS stores behind this iteration's loads.
        asm volatile("s_wait_dscnt 0" ::: "memory");

        fcol += 256;   // 32 pixels * 8 dims
        sqp  += 32;
        bp0  += 32;
        bp1  += 32;
    }

#pragma unroll
    for (int r = 0; r < 8; ++r) {
        const int n = n0 + r + 8 * lh;
        OUT[(size_t)n * 32 + ll]      = acc0[r];
        OUT[(size_t)n * 32 + ll + 16] = acc1[r];
    }
}

// ---------------------------------------------------------------------------
// Pointwise mean-field update: reduce K-split partials, normalize, 21x21
// mixing, compatibility, q = u - pairwise, softmax.  mode 0 = softmax(u).
// ---------------------------------------------------------------------------
__global__ __launch_bounds__(256) void pointwise_kernel(
    const float* __restrict__ u,       // [N][32]
    const float* __restrict__ filt,    // [2][KSPLIT][N][32]
    const float* __restrict__ sp_w, const float* __restrict__ sp_b,
    const float* __restrict__ bl_w, const float* __restrict__ bl_b,
    const float* __restrict__ cm_w, const float* __restrict__ cm_b,
    _Float16* __restrict__ sT,         // [32][N]
    float* __restrict__ out,           // [21][N]
    int mode, int last)
{
    int n = blockIdx.x * blockDim.x + threadIdx.x;
    if (n >= N_PIX) return;

    float q[NC];
    if (mode == 0) {
        for (int o = 0; o < NC; ++o) q[o] = u[(size_t)n * 32 + o];
    } else {
        // Fixed-order reduction of K-split partials (deterministic).
        float fs[NC + 1], fb[NC + 1];
        for (int c = 0; c <= NC; ++c) { fs[c] = 0.0f; fb[c] = 0.0f; }
        for (int s = 0; s < KSPLIT; ++s) {
            const float* ps = filt + ((size_t)(0 * KSPLIT + s) * N_PIX + n) * 32;
            const float* pb = filt + ((size_t)(1 * KSPLIT + s) * N_PIX + n) * 32;
            for (int c = 0; c <= NC; ++c) { fs[c] += ps[c]; fb[c] += pb[c]; }
        }
        float isp = 1.0f / fs[21];   // row normalizers from s-channel 21
        float ibl = 1.0f / fb[21];
        float mp[NC];
        for (int o = 0; o < NC; ++o) {
            float acc = sp_b[o] + bl_b[o];
            for (int c = 0; c < NC; ++c)
                acc += sp_w[o * NC + c] * (fs[c] * isp)
                     + bl_w[o * NC + c] * (fb[c] * ibl);
            mp[o] = acc;
        }
        for (int o = 0; o < NC; ++o) {
            float acc = cm_b[o];
            for (int c = 0; c < NC; ++c) acc += cm_w[o * NC + c] * mp[c];
            q[o] = u[(size_t)n * 32 + o] - acc;
        }
    }

    float mx = q[0];
    for (int o = 1; o < NC; ++o) mx = fmaxf(mx, q[o]);
    float e[NC], sum = 0.0f;
    for (int o = 0; o < NC; ++o) { e[o] = __expf(q[o] - mx); sum += e[o]; }
    float inv = 1.0f / sum;

    if (last) {
        for (int o = 0; o < NC; ++o) out[(size_t)o * N_PIX + n] = e[o] * inv;
    } else {
        for (int o = 0; o < NC; ++o)
            sT[(size_t)o * N_PIX + n] = (_Float16)(e[o] * inv);
        if (mode == 0) {
            sT[(size_t)21 * N_PIX + n] = (_Float16)1.0f;   // normalizer channel
#pragma unroll
            for (int o = 22; o < 32; ++o)
                sT[(size_t)o * N_PIX + n] = (_Float16)0.0f;
        }
    }
}

// ---------------------------------------------------------------------------
extern "C" void kernel_launch(void* const* d_in, const int* in_sizes, int n_in,
                              void* d_out, int out_size, void* d_ws, size_t ws_size,
                              hipStream_t stream) {
    const float* image = (const float*)d_in[0];
    const float* net_w = (const float*)d_in[1];
    const float* net_b = (const float*)d_in[2];
    const float* sp_w  = (const float*)d_in[3];
    const float* sp_b  = (const float*)d_in[4];
    const float* bl_w  = (const float*)d_in[5];
    const float* bl_b  = (const float*)d_in[6];
    const float* cm_w  = (const float*)d_in[7];
    const float* cm_b  = (const float*)d_in[8];
    float* out = (float*)d_out;

    char* ws = (char*)d_ws;
    size_t off = 0;
    auto carve = [&](size_t bytes) -> char* {
        off = (off + 255) & ~(size_t)255;
        char* p = ws + off;
        off += bytes;
        return p;
    };
    float*    u    = (float*)carve((size_t)N_PIX * 32 * sizeof(float));
    float*    feat = (float*)carve((size_t)2 * N_PIX * 8 * sizeof(float));
    float*    sq   = (float*)carve((size_t)2 * N_PIX * sizeof(float));
    float*    filt = (float*)carve((size_t)2 * KSPLIT * N_PIX * 32 * sizeof(float));
    _Float16* sT   = (_Float16*)carve((size_t)32 * N_PIX * sizeof(_Float16));
    (void)ws_size; (void)in_sizes; (void)n_in; (void)out_size;

    dim3 b256(256), gN((N_PIX + 255) / 256);
    unary_kernel<<<gN, b256, 0, stream>>>(image, net_w, net_b, u);
    feat_kernel<<<gN, b256, 0, stream>>>(image, feat, sq);
    pointwise_kernel<<<gN, b256, 0, stream>>>(u, filt, sp_w, sp_b, bl_w, bl_b,
                                              cm_w, cm_b, sT, out, 0, 0);
    for (int it = 0; it < 5; ++it) {
        filter_kernel<<<dim3(N_PIX / 16, 2, KSPLIT), dim3(32), 0, stream>>>(
            feat, sq, sT, filt);
        pointwise_kernel<<<gN, b256, 0, stream>>>(u, filt, sp_w, sp_b, bl_w, bl_b,
                                                  cm_w, cm_b, sT, out, 1,
                                                  (it == 4) ? 1 : 0);
    }
}
